// RegionProposalNetwork_42691974922361
// MI455X (gfx1250) — matrix operand order repaired and checked by reading, not compiled, not run on hardware
//
#include <hip/hip_runtime.h>
#include <hip/hip_bf16.h>

// ---------------------------------------------------------------------------
// Types for CDNA5 WMMA (wave32, 16x16x32 bf16 -> f32)
// ---------------------------------------------------------------------------
typedef __attribute__((ext_vector_type(16))) __bf16          v16bf;
typedef __attribute__((ext_vector_type(8)))  float           v8f;
typedef __attribute__((ext_vector_type(8)))  unsigned short  ushort8;

union FragU {
    ushort8 u8[2];
    v16bf   v;
};

__device__ __forceinline__ unsigned short f32_to_bf16_rne(float f) {
    unsigned int u = __float_as_uint(f);
    unsigned int r = 0x7FFFu + ((u >> 16) & 1u);
    return (unsigned short)((u + r) >> 16);
}

// ---------------------------------------------------------------------------
// Problem constants
// ---------------------------------------------------------------------------
#define NIMG   8
#define CIN    512
#define COUT   512
#define HF     64
#define WF     64
#define NPIX   (HF * WF)          // 4096
#define HP     66                  // padded (halo 1)
#define KTOT   (CIN * 9)           // 4608
#define NANCH  9
#define NBOX   (NPIX * NANCH)      // 36864
#define NPOST  300

// GEMM tiling: block 256 thr = 8 waves (2 M x 4 N), wave = 32x32 C (2x2 WMMA)
#define BM 64
#define BN 128
#define BK 32

// ---------------------------------------------------------------------------
// Kernel 1: pack x (f32 NCHW) -> bf16 with zero halo [n][ci][66][66]
// ---------------------------------------------------------------------------
__global__ __launch_bounds__(256)
void k_pack_x(const float* __restrict__ x, unsigned short* __restrict__ xpad) {
    int idx = blockIdx.x * 256 + threadIdx.x;
    const int total = NIMG * CIN * HP * HP;   // 17,842,176
    if (idx >= total) return;
    int xx  = idx % HP;
    int t1  = idx / HP;
    int yy  = t1 % HP;
    int t2  = t1 / HP;
    int ci  = t2 % CIN;
    int n   = t2 / CIN;
    float v = 0.0f;
    int y = yy - 1, xq = xx - 1;
    if (y >= 0 && y < HF && xq >= 0 && xq < WF) {
        v = x[(((size_t)n * CIN + ci) * HF + y) * WF + xq];
    }
    xpad[idx] = f32_to_bf16_rne(v);
}

// ---------------------------------------------------------------------------
// Kernel 2: pack W1 (f32) -> bf16, same [co][k] layout (k = ci*9+ky*3+kx).
// This IS the B "[n][k]" layout the WMMA B-fragment wants: pure copy-convert.
// ---------------------------------------------------------------------------
__global__ __launch_bounds__(256)
void k_pack_w(const float* __restrict__ W1, unsigned short* __restrict__ w1b) {
    int idx = blockIdx.x * 256 + threadIdx.x;
    const int total = COUT * KTOT;            // 2,359,296
    if (idx >= total) return;
    w1b[idx] = f32_to_bf16_rne(W1[idx]);
}

// ---------------------------------------------------------------------------
// Kernel 3: implicit-im2col GEMM conv via v_wmma_f32_16x16x32_bf16
//   C[pix][co] = sum_k A[pix][k] * B[co][k];  k = ci*9 + (ky*3+kx)
//   Double-buffered LDS, 4 WMMAs per wave per K-step, 144 K-steps.
// ---------------------------------------------------------------------------
__global__ __launch_bounds__(256)
void k_conv_wmma(const unsigned short* __restrict__ xpad,
                 const unsigned short* __restrict__ w1b,
                 const float* __restrict__ b1,
                 float* __restrict__ h1) {
    __shared__ unsigned short As[2][BM][BK];   // [m][k]   8 KB
    __shared__ unsigned short Bs[2][BN][BK];   // [n][k]  16 KB

    const int n_img = blockIdx.z;
    const int tm    = blockIdx.x;     // pixel tile 0..63
    const int tn    = blockIdx.y;     // channel tile 0..3
    const int tid   = threadIdx.x;
    const int lane  = tid & 31;
    const int wave  = tid >> 5;       // 0..7
    const int wm    = (wave >> 2) * 32;   // 0 | 32
    const int wn    = (wave & 3) * 32;    // 0 | 32 | 64 | 96

    const unsigned short* xp = xpad + (size_t)n_img * CIN * HP * HP;

    // per-lane fragment coordinates (ISA 7.12.2, 16-bit A 16x32 / B 32x16)
    const int a_m  = lane & 15;
    const int a_kb = (lane >> 4) * 8;
    const int b_n  = lane & 15;
    const int b_k0 = (lane >> 4) * 8;

    // A-tile loader: thread t -> m = t/4 (0..63), kbase = (t%4)*8
    const int lm    = tid >> 2;
    const int lkb   = (tid & 3) * 8;
    const int pixA  = tm * BM + lm;
    const int pyA   = pixA >> 6;
    const int pxA   = pixA & 63;

    // B-tile loader: thread t -> n = t/2 (0..127), koff = (t%2)*16 (contiguous k)
    const int bn_l  = tid >> 1;
    const int bk_l  = (tid & 1) * 16;
    const unsigned short* wrow = w1b + (size_t)(tn * BN + bn_l) * KTOT + bk_l;

    v8f acc00 = {}, acc01 = {}, acc10 = {}, acc11 = {};

    auto load_tile = [&](int k0, int buf) {
        // A: scattered im2col gather (halo-padded, branch-free)
        #pragma unroll
        for (int j = 0; j < 8; ++j) {
            int k   = k0 + lkb + j;
            int ci  = k / 9;
            int off = k - ci * 9;
            int ky  = off / 3;
            int kx  = off - ky * 3;
            As[buf][lm][lkb + j] =
                xp[(size_t)ci * (HP * HP) + (size_t)(pyA + ky) * HP + (pxA + kx)];
        }
        // B: two 16B vector loads, contiguous along k; no transpose needed
        const ushort8* src = (const ushort8*)(wrow + k0);
        *(ushort8*)&Bs[buf][bn_l][bk_l]     = src[0];
        *(ushort8*)&Bs[buf][bn_l][bk_l + 8] = src[1];
        if (k0 + BK < KTOT) {
            __builtin_prefetch(wrow + k0 + BK, 0, 1);
        }
    };

    auto compute = [&](int buf) {
        FragU a0, a1, bb0, bb1;
        a0.u8[0]  = *(const ushort8*)&As[buf][wm + a_m][a_kb];
        a0.u8[1]  = *(const ushort8*)&As[buf][wm + a_m][16 + a_kb];
        a1.u8[0]  = *(const ushort8*)&As[buf][wm + 16 + a_m][a_kb];
        a1.u8[1]  = *(const ushort8*)&As[buf][wm + 16 + a_m][16 + a_kb];
        bb0.u8[0] = *(const ushort8*)&Bs[buf][wn + b_n][b_k0];
        bb0.u8[1] = *(const ushort8*)&Bs[buf][wn + b_n][16 + b_k0];
        bb1.u8[0] = *(const ushort8*)&Bs[buf][wn + 16 + b_n][b_k0];
        bb1.u8[1] = *(const ushort8*)&Bs[buf][wn + 16 + b_n][16 + b_k0];
        acc00 = __builtin_amdgcn_wmma_f32_16x16x32_bf16(
            false, a0.v, false, bb0.v, (short)0, acc00, false, false);
        acc01 = __builtin_amdgcn_wmma_f32_16x16x32_bf16(
            false, a0.v, false, bb1.v, (short)0, acc01, false, false);
        acc10 = __builtin_amdgcn_wmma_f32_16x16x32_bf16(
            false, a1.v, false, bb0.v, (short)0, acc10, false, false);
        acc11 = __builtin_amdgcn_wmma_f32_16x16x32_bf16(
            false, a1.v, false, bb1.v, (short)0, acc11, false, false);
    };

    load_tile(0, 0);
    int buf = 0;
    for (int k0 = 0; k0 < KTOT; k0 += BK) {
        __syncthreads();
        if (k0 + BK < KTOT) load_tile(k0 + BK, buf ^ 1);
        compute(buf);
        buf ^= 1;
    }

    // ---- C writeback with bias + ReLU; C layout: N=lane&15, M=r+(lane>>4)*8
    const int cbase = tn * BN + wn + (lane & 15);
    const int mbase = tm * BM + wm + ((lane >> 4) * 8);
    const float bv0 = b1[cbase];
    const float bv1 = b1[cbase + 16];
    float* hb = h1 + (size_t)n_img * NPIX * COUT;
    #pragma unroll
    for (int r = 0; r < 8; ++r) {
        int m0 = mbase + r;
        int m1 = mbase + 16 + r;
        hb[(size_t)m0 * COUT + cbase]      = fmaxf(acc00[r] + bv0, 0.0f);
        hb[(size_t)m0 * COUT + cbase + 16] = fmaxf(acc01[r] + bv1, 0.0f);
        hb[(size_t)m1 * COUT + cbase]      = fmaxf(acc10[r] + bv0, 0.0f);
        hb[(size_t)m1 * COUT + cbase + 16] = fmaxf(acc11[r] + bv1, 0.0f);
    }
}

// ---------------------------------------------------------------------------
// Kernel 4: 1x1 heads.  Per pixel: 36 loc outputs + 18 score outputs,
// each a 512-length dot; then fg = sigmoid(s1 - s0) per anchor.
// ---------------------------------------------------------------------------
__global__ __launch_bounds__(64)
void k_heads(const float* __restrict__ h1,
             const float* __restrict__ Wsc, const float* __restrict__ bsc,
             const float* __restrict__ Wlo, const float* __restrict__ blo,
             float* __restrict__ out_locs, float* __restrict__ out_fg) {
    __shared__ float hrow[CIN];
    __shared__ float sres[54];
    const int gid   = blockIdx.x;          // 0..32767
    const int n_img = gid >> 12;
    const int pix   = gid & (NPIX - 1);
    const float* h  = h1 + (size_t)gid * CIN;
    for (int i = threadIdx.x; i < CIN; i += 64) hrow[i] = h[i];
    __syncthreads();

    const int o = threadIdx.x;
    if (o < 54) {
        const float* wr;
        float s;
        if (o < 36) { wr = Wlo + (size_t)o * CIN;        s = blo[o]; }
        else        { wr = Wsc + (size_t)(o - 36) * CIN; s = bsc[o - 36]; }
        #pragma unroll 4
        for (int i = 0; i < CIN; ++i) s += hrow[i] * wr[i];
        sres[o] = s;
    }
    __syncthreads();

    if (o < 36) {
        int a = o >> 2, d = o & 3;
        out_locs[(((size_t)n_img * NBOX) + (size_t)pix * NANCH + a) * 4 + d] = sres[o];
    }
    if (o < NANCH) {
        float s0 = sres[36 + o * 2];
        float s1 = sres[36 + o * 2 + 1];
        out_fg[(size_t)n_img * NBOX + (size_t)pix * NANCH + o] =
            1.0f / (1.0f + __expf(s0 - s1));
    }
}

// ---------------------------------------------------------------------------
// Kernel 5: proposals. One block per image: loc2bbox + clip + min-size
// filter, then 300 rounds of argmax-NMS (block reduction in LDS).
// ---------------------------------------------------------------------------
__global__ __launch_bounds__(1024)
void k_proposals(const float* __restrict__ locs, const float* __restrict__ fg,
                 float* __restrict__ boxes, float* __restrict__ scores,
                 float* __restrict__ rois, int* __restrict__ roi_idx,
                 const int* __restrict__ img_size_p) {
    const int n_img = blockIdx.x;
    const int tid   = threadIdx.x;
    __shared__ float rv[1024];
    __shared__ int   ri[1024];
    __shared__ float bb[4];

    const float img = (float)(*img_size_p);
    float*       bxp = boxes  + (size_t)n_img * NBOX * 4;
    float*       scp = scores + (size_t)n_img * NBOX;
    const float* lo  = locs   + (size_t)n_img * NBOX * 4;
    const float* fgp = fg     + (size_t)n_img * NBOX;
    const float NEGINF = -__builtin_inff();

    // --- build boxes & filtered scores ---
    for (int i = tid; i < NBOX; i += 1024) {
        int pix = i / 9;
        int a   = i - pix * 9;
        int py  = pix >> 6, px = pix & 63;
        int rix = a / 3, six = a - rix * 3;
        float r = (rix == 0) ? 0.5f : (rix == 1 ? 1.0f : 2.0f);
        float s = (six == 0) ? 8.0f : (six == 1 ? 16.0f : 32.0f);
        float ah = 16.0f * s * __fsqrt_rn(r);
        float aw = 16.0f * s * __fsqrt_rn(1.0f / r);
        float cy = 8.0f + py * 16.0f;
        float cx = 8.0f + px * 16.0f;
        float dy = lo[i * 4 + 0], dx = lo[i * 4 + 1];
        float dh = lo[i * 4 + 2], dw = lo[i * 4 + 3];
        float ncy = dy * ah + cy;
        float ncx = dx * aw + cx;
        float nh  = __expf(dh) * ah;
        float nw  = __expf(dw) * aw;
        float y0 = fminf(fmaxf(ncy - 0.5f * nh, 0.0f), img);
        float x0 = fminf(fmaxf(ncx - 0.5f * nw, 0.0f), img);
        float y1 = fminf(fmaxf(ncy + 0.5f * nh, 0.0f), img);
        float x1 = fminf(fmaxf(ncx + 0.5f * nw, 0.0f), img);
        bxp[i * 4 + 0] = y0; bxp[i * 4 + 1] = x0;
        bxp[i * 4 + 2] = y1; bxp[i * 4 + 3] = x1;
        float hs = y1 - y0, ws = x1 - x0;
        scp[i] = (hs >= 16.0f && ws >= 16.0f) ? fgp[i] : NEGINF;
    }
    __syncthreads();

    // --- greedy NMS, 300 picks ---
    for (int it = 0; it < NPOST; ++it) {
        float bestv = NEGINF; int besti = 0;
        for (int i = tid; i < NBOX; i += 1024) {
            float v = scp[i];
            if (v > bestv) { bestv = v; besti = i; }
        }
        rv[tid] = bestv; ri[tid] = besti;
        __syncthreads();
        for (int sN = 512; sN > 0; sN >>= 1) {
            if (tid < sN && rv[tid + sN] > rv[tid]) {
                rv[tid] = rv[tid + sN]; ri[tid] = ri[tid + sN];
            }
            __syncthreads();
        }
        if (tid == 0) {
            int j = ri[0];
            bb[0] = bxp[j * 4 + 0]; bb[1] = bxp[j * 4 + 1];
            bb[2] = bxp[j * 4 + 2]; bb[3] = bxp[j * 4 + 3];
            size_t o = ((size_t)n_img * NPOST + it) * 4;
            rois[o + 0] = bb[0]; rois[o + 1] = bb[1];
            rois[o + 2] = bb[2]; rois[o + 3] = bb[3];
            roi_idx[n_img * NPOST + it] = n_img;
            scp[j] = NEGINF;
        }
        __syncthreads();
        float by0 = bb[0], bx0 = bb[1], by1 = bb[2], bx1 = bb[3];
        float barea = (by1 - by0) * (bx1 - bx0);
        for (int i = tid; i < NBOX; i += 1024) {
            float y0 = bxp[i * 4 + 0], x0 = bxp[i * 4 + 1];
            float y1 = bxp[i * 4 + 2], x1 = bxp[i * 4 + 3];
            float ty = fmaxf(by0, y0), tx = fmaxf(bx0, x0);
            float by = fminf(by1, y1), bx2 = fminf(bx1, x1);
            float inter = fmaxf(by - ty, 0.0f) * fmaxf(bx2 - tx, 0.0f);
            float a2 = (y1 - y0) * (x1 - x0);
            float iou = inter / (barea + a2 - inter + 1e-9f);
            if (iou > 0.7f) scp[i] = NEGINF;
        }
        __syncthreads();
    }
}

// ---------------------------------------------------------------------------
// Host side
// ---------------------------------------------------------------------------
extern "C" void kernel_launch(void* const* d_in, const int* in_sizes, int n_in,
                              void* d_out, int out_size, void* d_ws, size_t ws_size,
                              hipStream_t stream) {
    const float* x       = (const float*)d_in[0];   // 8*512*64*64
    const float* W1      = (const float*)d_in[1];   // 512*512*3*3
    const float* b1      = (const float*)d_in[2];   // 512
    const float* W_score = (const float*)d_in[3];   // 18*512
    const float* b_score = (const float*)d_in[4];   // 18
    const float* W_loc   = (const float*)d_in[5];   // 36*512
    const float* b_loc   = (const float*)d_in[6];   // 36
    const int*   img_sz  = (const int*)d_in[7];     // scalar 1024

    // workspace layout (all 64B aligned)
    unsigned char* ws = (unsigned char*)d_ws;
    unsigned short* xpad = (unsigned short*)ws;                        // 35,684,352 B
    unsigned short* w1b  = (unsigned short*)(ws + 35684352u);          //  4,718,592 B
    float* h1            = (float*)(ws + 40402944u);                   // 67,108,864 B
    float* boxes         = (float*)(ws + 107511808u);                  //  4,718,592 B
    float* scores        = (float*)(ws + 112230400u);                  //  1,179,648 B

    // d_out layout: rpn_locs | rpn_fg | rois | roi_indices
    float* out_locs = (float*)d_out;                      // 1,179,648
    float* out_fg   = out_locs + (size_t)NIMG * NBOX * 4; //   294,912
    float* out_rois = out_fg + (size_t)NIMG * NBOX;       //     9,600
    int*   out_idx  = (int*)(out_rois + (size_t)NIMG * NPOST * 4); // 2,400

    // 1. pack input with halo
    {
        int total = NIMG * CIN * HP * HP;
        k_pack_x<<<(total + 255) / 256, 256, 0, stream>>>(x, xpad);
    }
    // 2. pack weights (copy-convert, [co][k] already matches B layout)
    {
        int total = COUT * KTOT;
        k_pack_w<<<(total + 255) / 256, 256, 0, stream>>>(W1, w1b);
    }
    // 3. WMMA conv + bias + relu -> h1 [n][pix][co]
    {
        dim3 grid(NPIX / BM, COUT / BN, NIMG);   // 64 x 4 x 8
        k_conv_wmma<<<grid, 256, 0, stream>>>(xpad, w1b, b1, h1);
    }
    // 4. heads: locs + fg
    k_heads<<<NIMG * NPIX, 64, 0, stream>>>(h1, W_score, b_score, W_loc, b_loc,
                                            out_locs, out_fg);
    // 5. proposals: loc2bbox + clip + filter + NMS -> rois, roi_indices
    k_proposals<<<NIMG, 1024, 0, stream>>>(out_locs, out_fg, boxes, scores,
                                           out_rois, out_idx, img_sz);
}
